// NeoAttention_6588479832078
// MI455X (gfx1250) — compile-verified, matrix operand order checked by reading, not run on hardware
//
#include <hip/hip_runtime.h>
#include <math.h>

// ---------------- problem constants ----------------
#define Bc   2
#define Sc   2048
#define HIDc 2048
#define Hc   16
#define KVc  8
#define HDc  128
#define D2c  256          // concat(qt[128], qh[64], qw[64])
#define WINc 1024
#define EPSc 1e-6f
#define THETA_T  1000000.0f
#define THETA_HW 10000.0f
#define SCALEc   0.08838834764831845f   // 128^-0.5

typedef __attribute__((ext_vector_type(16))) __bf16 v16bf;
typedef __attribute__((ext_vector_type(8)))  float  v8f;

#define WMMA_BF16(a, b, c) \
  __builtin_amdgcn_wmma_f32_16x16x32_bf16(false, (a), false, (b), (short)0, (c), false, false)

// K index pattern for the 16-bit A-matrix 16x32 layout (ISA 7.12.2):
// lanes 0-15 hold K {0..7,16..23}, lanes 16-31 hold K {8..15,24..31}
__device__ __forceinline__ int kidx16(int lane, int j) {
  int half = (lane & 16) ? 8 : 0;
  return ((j >> 3) << 4) + half + (j & 7);
}

__device__ __forceinline__ v8f zero8() {
  v8f z;
#pragma unroll
  for (int i = 0; i < 8; ++i) z[i] = 0.0f;
  return z;
}

// ---- CDNA5 async global->LDS copy (ASYNCcnt path, ISA 10 / 15.18.3) ----
// per-lane: LDS[lds_byte_off] = MEM[gsrc], 8 bytes
__device__ __forceinline__ void async_load_b64_to_lds(const __bf16* gsrc, unsigned lds_off) {
  unsigned long long ga = (unsigned long long)(size_t)gsrc;
  asm volatile("global_load_async_to_lds_b64 %0, %1, off"
               :: "v"(lds_off), "v"(ga)
               : "memory");
}
__device__ __forceinline__ void wait_async0() {
  asm volatile("s_wait_asynccnt 0x0" ::: "memory");
}

// A fragment: row-major A[M,K], this lane owns row (base + lane%16)
__device__ __forceinline__ v16bf frag_a_rm(const __bf16* A, int ld, int row, int k0, int lane) {
  v16bf f;
#pragma unroll
  for (int j = 0; j < 16; ++j) f[j] = A[(size_t)row * ld + (size_t)(k0 + kidx16(lane, j))];
  return f;
}

// ---------------- kernel 1: f32 -> bf16 cast ----------------
__global__ void cast_f32_bf16(const float* __restrict__ src, __bf16* __restrict__ dst, int n) {
  int i = blockIdx.x * blockDim.x + threadIdx.x;
  if (i < n) dst[i] = (__bf16)src[i];
}

// ---------------- kernel 2: bf16 WMMA GEMM with async-LDS staged B ----------------
// C[M,N] = A[M,K] @ B[K,N]; 4 waves/block, each wave one 16x16 tile (64x16 per block).
// The shared 32(K)x16(N) B tile is staged once per k-step via GLOBAL_LOAD_ASYNC_TO_LDS_B64
// and consumed by all 4 waves from LDS. Requires M%64==0, N%16==0, K%32==0.
__global__ void gemm_bf16(const __bf16* __restrict__ A, const __bf16* __restrict__ Bm,
                          void* __restrict__ Cout, int M, int N, int K, int c_is_f32) {
  __shared__ __bf16 sB[32 * 16];   // [k][n] row-major, 1 KB

  int tid  = threadIdx.x;
  int lane = tid & 31;
  int wave = tid >> 5;
  int n0 = blockIdx.x * 16;
  int m0 = (blockIdx.y * 4 + wave) * 16;

  int row = m0 + (lane & 15);
  int col = lane & 15;            // column within the 16-wide N tile
  v8f acc = zero8();

  // staging pattern: thread t copies 8B (4 bf16): B row = t>>2, col group = (t&3)*4
  int brow  = tid >> 2;
  int bcol4 = (tid & 3) * 4;
  unsigned sB_off = (unsigned)(size_t)(&sB[brow * 16 + bcol4]);

  for (int k0 = 0; k0 < K; k0 += 32) {
    async_load_b64_to_lds(Bm + (size_t)(k0 + brow) * N + n0 + bcol4, sB_off);
    wait_async0();                 // this wave's async copies landed in LDS
    __syncthreads();               // all waves' copies visible

    __builtin_prefetch(&A[(size_t)row * K + k0 + 256], 0, 1);  // global_prefetch_b8
    v16bf a = frag_a_rm(A, K, row, k0, lane);
    v16bf b;
#pragma unroll
    for (int j = 0; j < 16; ++j) b[j] = sB[kidx16(lane, j) * 16 + col];
    acc = WMMA_BF16(a, b, acc);

    __syncthreads();               // done reading sB before next-iteration overwrite
  }

  int g = lane >> 4;  // C/D layout: vgpr v holds row v + 8*g
  if (c_is_f32) {
    float* C = (float*)Cout;
#pragma unroll
    for (int v = 0; v < 8; ++v) C[(size_t)(m0 + v + 8 * g) * N + n0 + col] = acc[v];
  } else {
    __bf16* C = (__bf16*)Cout;
#pragma unroll
    for (int v = 0; v < 8; ++v) C[(size_t)(m0 + v + 8 * g) * N + n0 + col] = (__bf16)acc[v];
  }
}

// ---------------- kernel 3: RMS-norm + 3-axis partial RoPE ----------------
// tproj/hwproj: [B,S,nH,HD] bf16 ; out cat: [B,nH,S,256] bf16
__global__ void rmsrope(const __bf16* __restrict__ tproj, const __bf16* __restrict__ hwproj,
                        const int* __restrict__ idx,
                        const float* __restrict__ nt, const float* __restrict__ nh,
                        const float* __restrict__ nw,
                        __bf16* __restrict__ cat, int nH) {
  int bid = blockIdx.x;          // ((b*S + s)*nH + h)
  int h  = bid % nH;
  int bs = bid / nH;
  int s  = bs % Sc;
  int b  = bs / Sc;
  int e  = threadIdx.x;          // 0..127

  float xt  = (float)tproj [(size_t)bid * HDc + e];
  float xhw = (float)hwproj[(size_t)bid * HDc + e];

  __shared__ float sh[HDc];
  __shared__ float sums[3];
  __shared__ float xt_n[HDc];
  __shared__ float xhw_n[HDc];

  sh[e] = xt * xt;
  __syncthreads();
  if (e == 0) { float a = 0.f; for (int i = 0; i < 128; ++i) a += sh[i]; sums[0] = a; }
  __syncthreads();
  sh[e] = xhw * xhw;
  __syncthreads();
  if (e == 0) {
    float a = 0.f; for (int i = 0; i < 64; ++i)  a += sh[i];      sums[1] = a;
    a = 0.f;       for (int i = 64; i < 128; ++i) a += sh[i];     sums[2] = a;
  }
  __syncthreads();

  float rt = rsqrtf(sums[0] / 128.0f + EPSc);
  float rh = rsqrtf(sums[1] / 64.0f + EPSc);
  float rw = rsqrtf(sums[2] / 64.0f + EPSc);

  xt_n[e]  = xt * rt * nt[e];
  xhw_n[e] = xhw * ((e < 64) ? rh * nh[e] : rw * nw[e - 64]);
  __syncthreads();

  size_t base = ((size_t)(b * nH + h) * Sc + s) * D2c;

  // t axis: dim 128, theta 1e6, pos = indexes[0][s]
  {
    float pos = (float)idx[0 * Sc + s];
    float fr  = powf(THETA_T, -2.0f * (float)(e & 63) / 128.0f);
    float ang = pos * fr;
    float rot = (e < 64) ? -xt_n[e + 64] : xt_n[e - 64];
    cat[base + e] = (__bf16)(xt_n[e] * cosf(ang) + rot * sinf(ang));
  }
  // h/w axes: dim 64, theta 1e4, pos = indexes[1][s] / indexes[2][s]
  {
    int j2 = e & 63;
    float pos = (e < 64) ? (float)idx[Sc + s] : (float)idx[2 * Sc + s];
    float fr  = powf(THETA_HW, -2.0f * (float)(j2 & 31) / 64.0f);
    float ang = pos * fr;
    int   hb  = (e < 64) ? 0 : 64;
    float rot = (j2 < 32) ? -xhw_n[hb + j2 + 32] : xhw_n[hb + j2 - 32];
    cat[base + 128 + e] = (__bf16)(xhw_n[e] * cosf(ang) + rot * sinf(ang));
  }
}

// ---------------- kernel 4: sliding-window flash attention ----------------
// qcat [B,H,S,256], kcat [B,KV,S,256], vproj [B,S,KV,HD] ; aout [B,S,H*HD] bf16
// One wave (32 threads) per (b,h, 16-query tile). Online softmax over 32-key blocks.
__global__ void attn(const __bf16* __restrict__ qcat, const __bf16* __restrict__ kcat,
                     const __bf16* __restrict__ vproj, __bf16* __restrict__ aout) {
  int bh = blockIdx.x;
  int h  = bh % Hc;
  int b  = bh / Hc;
  int kv = h / (Hc / KVc);
  int q0 = blockIdx.y * 16;
  int lane = threadIdx.x;
  int g = lane >> 4;
  int cl = lane & 15;

  const __bf16* Q  = qcat + (size_t)(b * Hc + h) * Sc * D2c;    // [S,256]
  const __bf16* Kc = kcat + (size_t)(b * KVc + kv) * Sc * D2c;  // [S,256]

  __shared__ float  sS[16 * 32];
  __shared__ __bf16 sP[16 * 32];
  __shared__ float  sAlpha[16];
  __shared__ float  sL[16];

  float m_r = -1e30f, l_r = 0.0f;  // meaningful in lanes 0..15 (row = lane)
  v8f o[8];
#pragma unroll
  for (int i = 0; i < 8; ++i) o[i] = zero8();

  int kstart = q0 - (WINc - 1);
  if (kstart < 0) kstart = 0;
  kstart &= ~31;
  int kend = q0 + 16;  // exclusive on k0; block covers keys [k0, k0+31]

  int qrow = q0 + cl;

  for (int k0 = kstart; k0 < kend; k0 += 32) {
    // ---- scores S = Q K^T over d=256, two 16-key sub-tiles ----
    v8f s0 = zero8(), s1 = zero8();
    for (int d0 = 0; d0 < D2c; d0 += 32) {
      v16bf a = frag_a_rm(Q, D2c, qrow, d0, lane);
      v16bf b0, b1;
#pragma unroll
      for (int j = 0; j < 16; ++j) {
        int kd = d0 + kidx16(lane, j);
        b0[j] = Kc[(size_t)(k0 + cl) * D2c + kd];        // B = K^T (col = key)
        b1[j] = Kc[(size_t)(k0 + 16 + cl) * D2c + kd];
      }
      s0 = WMMA_BF16(a, b0, s0);
      s1 = WMMA_BF16(a, b1, s1);
    }
#pragma unroll
    for (int v = 0; v < 8; ++v) {
      sS[(v + 8 * g) * 32 + cl]      = s0[v];
      sS[(v + 8 * g) * 32 + 16 + cl] = s1[v];
    }
    __syncthreads();

    // ---- online softmax (lanes 0..15 each own one query row) ----
    if (lane < 16) {
      int r = lane;
      int qq = q0 + r;
      float sc[32];
      float mx = m_r;
#pragma unroll
      for (int c = 0; c < 32; ++c) {
        int kk = k0 + c;
        int diff = qq - kk;
        float sv = (diff >= 0 && diff < WINc) ? sS[r * 32 + c] * SCALEc : -1e30f;
        sc[c] = sv;
        if (sv > mx) mx = sv;
      }
      float alpha = __expf(m_r - mx);
      float psum = 0.0f;
#pragma unroll
      for (int c = 0; c < 32; ++c) {
        float p = (sc[c] > -1e29f) ? __expf(sc[c] - mx) : 0.0f;  // robust to all-masked blocks
        psum += p;
        sP[r * 32 + c] = (__bf16)p;
      }
      m_r = mx;
      l_r = l_r * alpha + psum;
      sAlpha[r] = alpha;
      sL[r] = l_r;
    }
    __syncthreads();

    // ---- rescale O, then O += P @ V (16x128 via 8 column tiles) ----
    float alr[8];
#pragma unroll
    for (int v = 0; v < 8; ++v) alr[v] = sAlpha[v + 8 * g];
#pragma unroll
    for (int nt = 0; nt < 8; ++nt)
#pragma unroll
      for (int v = 0; v < 8; ++v) o[nt][v] *= alr[v];

    v16bf pA;
#pragma unroll
    for (int j = 0; j < 16; ++j) pA[j] = sP[cl * 32 + kidx16(lane, j)];

    for (int nt = 0; nt < 8; ++nt) {
      v16bf vB;
#pragma unroll
      for (int j = 0; j < 16; ++j) {
        int kk = k0 + kidx16(lane, j);
        vB[j] = vproj[((size_t)(b * Sc + kk) * KVc + kv) * HDc + nt * 16 + cl];
      }
      o[nt] = WMMA_BF16(pA, vB, o[nt]);
    }
    __syncthreads();
  }

  // ---- finalize: divide by l, write [B,S,H*HD] bf16 ----
  float lr[8];
#pragma unroll
  for (int v = 0; v < 8; ++v) {
    float L = sL[v + 8 * g];
    lr[v] = (L > 0.0f) ? 1.0f / L : 0.0f;
  }
#pragma unroll
  for (int nt = 0; nt < 8; ++nt)
#pragma unroll
    for (int v = 0; v < 8; ++v) {
      int row = q0 + v + 8 * g;
      aout[(size_t)(b * Sc + row) * (Hc * HDc) + h * HDc + nt * 16 + cl] =
          (__bf16)(o[nt][v] * lr[v]);
    }
}

// ---------------- host orchestration ----------------
extern "C" void kernel_launch(void* const* d_in, const int* in_sizes, int n_in,
                              void* d_out, int out_size, void* d_ws, size_t ws_size,
                              hipStream_t stream) {
  const float* hs   = (const float*)d_in[0];
  const int*   idx  = (const int*)d_in[1];
  const float* wq   = (const float*)d_in[2];
  const float* wqhw = (const float*)d_in[3];
  const float* wk   = (const float*)d_in[4];
  const float* wkhw = (const float*)d_in[5];
  const float* wv   = (const float*)d_in[6];
  const float* wo   = (const float*)d_in[7];
  const float* qn_t = (const float*)d_in[8];
  const float* qn_h = (const float*)d_in[9];
  const float* qn_w = (const float*)d_in[10];
  const float* kn_t = (const float*)d_in[11];
  const float* kn_h = (const float*)d_in[12];
  const float* kn_w = (const float*)d_in[13];
  (void)in_sizes; (void)n_in; (void)out_size; (void)ws_size;

  const int M    = Bc * Sc;       // 4096 tokens
  const int Nq   = Hc * HDc;      // 2048
  const int Nkv  = KVc * HDc;     // 1024
  const int K    = HIDc;          // 2048

  char* ws = (char*)d_ws;
  size_t off = 0;
  auto carve = [&](size_t bytes) -> void* {
    void* p = ws + off;
    off = (off + bytes + 255) & ~(size_t)255;
    return p;
  };

  __bf16* hsb    = (__bf16*)carve((size_t)M * K * 2);
  __bf16* wqb    = (__bf16*)carve((size_t)K * Nq * 2);
  __bf16* wqhwb  = (__bf16*)carve((size_t)K * Nq * 2);
  __bf16* wkb    = (__bf16*)carve((size_t)K * Nkv * 2);
  __bf16* wkhwb  = (__bf16*)carve((size_t)K * Nkv * 2);
  __bf16* wvb    = (__bf16*)carve((size_t)K * Nkv * 2);
  __bf16* wob    = (__bf16*)carve((size_t)Nq * K * 2);
  __bf16* qtp    = (__bf16*)carve((size_t)M * Nq * 2);
  __bf16* qhwp   = (__bf16*)carve((size_t)M * Nq * 2);
  __bf16* ktp    = (__bf16*)carve((size_t)M * Nkv * 2);
  __bf16* khwp   = (__bf16*)carve((size_t)M * Nkv * 2);
  __bf16* vp     = (__bf16*)carve((size_t)M * Nkv * 2);
  __bf16* q_cat  = (__bf16*)carve((size_t)Bc * Hc * Sc * D2c * 2);
  __bf16* k_cat  = (__bf16*)carve((size_t)Bc * KVc * Sc * D2c * 2);
  __bf16* a_out  = (__bf16*)carve((size_t)M * Nq * 2);

  auto cast = [&](const float* s, __bf16* d, int n) {
    cast_f32_bf16<<<dim3((n + 255) / 256), dim3(256), 0, stream>>>(s, d, n);
  };
  cast(hs,   hsb,   M * K);
  cast(wq,   wqb,   K * Nq);
  cast(wqhw, wqhwb, K * Nq);
  cast(wk,   wkb,   K * Nkv);
  cast(wkhw, wkhwb, K * Nkv);
  cast(wv,   wvb,   K * Nkv);
  cast(wo,   wob,   Nq * K);

  // projections (bf16 out)
  gemm_bf16<<<dim3(Nq / 16,  M / 64), dim3(128), 0, stream>>>(hsb, wqb,   qtp,  M, Nq,  K, 0);
  gemm_bf16<<<dim3(Nq / 16,  M / 64), dim3(128), 0, stream>>>(hsb, wqhwb, qhwp, M, Nq,  K, 0);
  gemm_bf16<<<dim3(Nkv / 16, M / 64), dim3(128), 0, stream>>>(hsb, wkb,   ktp,  M, Nkv, K, 0);
  gemm_bf16<<<dim3(Nkv / 16, M / 64), dim3(128), 0, stream>>>(hsb, wkhwb, khwp, M, Nkv, K, 0);
  gemm_bf16<<<dim3(Nkv / 16, M / 64), dim3(128), 0, stream>>>(hsb, wvb,   vp,   M, Nkv, K, 0);

  // RMS + 3-axis RoPE -> q_cat / k_cat
  rmsrope<<<dim3(Bc * Sc * Hc),  dim3(128), 0, stream>>>(qtp, qhwp, idx, qn_t, qn_h, qn_w, q_cat, Hc);
  rmsrope<<<dim3(Bc * Sc * KVc), dim3(128), 0, stream>>>(ktp, khwp, idx, kn_t, kn_h, kn_w, k_cat, KVc);

  // sliding-window attention
  attn<<<dim3(Bc * Hc, Sc / 16), dim3(32), 0, stream>>>(q_cat, k_cat, vp, a_out);

  // output projection (f32 out -> d_out)
  gemm_bf16<<<dim3(Nq / 16, M / 64), dim3(128), 0, stream>>>(a_out, wob, (float*)d_out, M, Nq, K, 1);
}
